// PolyphonicSubspaceAttentionV20_47648367182306
// MI455X (gfx1250) — compile-verified, matrix operand order checked by reading, not run on hardware
//
#include <hip/hip_runtime.h>
#include <hip/hip_bf16.h>

typedef unsigned short u16;
typedef unsigned int   u32;
typedef unsigned long long u64;

typedef __attribute__((ext_vector_type(16))) __bf16 v16bf;
typedef __attribute__((ext_vector_type(8)))  float  v8f;

#define BDIM 1024   // D
#define LDIM 1024   // L
#define BB   4      // B
#define HH   16     // H
#define HD   64     // head dim
#define ROWS (BB*LDIM)      // 4096
#define NQKV (3*BDIM)       // 3072

__device__ __forceinline__ u16 f2bf(float f) {
    union { float f; u32 u; } x; x.f = f;
    u32 r = x.u + 0x7FFFu + ((x.u >> 16) & 1u);
    return (u16)(r >> 16);
}

struct Frag { union { v16bf v; uint4 q[2]; u16 h[16]; }; };

// load a 16-half fragment (A-layout) from a row:
// lane holds row (lane&15); halves 0..7 at K = g*8.. , halves 8..15 at K = 16+g*8..
__device__ __forceinline__ void load_frag_u16(Frag& f, const u16* rowPtr, int g) {
    f.q[0] = *(const uint4*)(rowPtr + g * 8);
    f.q[1] = *(const uint4*)(rowPtr + 16 + g * 8);
}

// async copy of one 16B chunk from global to LDS (ASYNCcnt-tracked, no VGPR data)
__device__ __forceinline__ void async_copy16(u32 ldsAddr, const void* gaddr) {
    asm volatile("global_load_async_to_lds_b128 %0, %1, off"
                 :: "v"(ldsAddr), "v"((u64)(uintptr_t)gaddr) : "memory");
}
__device__ __forceinline__ void wait_async0() {
    asm volatile("s_wait_asynccnt 0x0" ::: "memory");
}

// ---------------------------------------------------------------- prep kernels
__global__ __launch_bounds__(256) void cvt_x_kernel(const float* __restrict__ src,
                                                    u16* __restrict__ dst, int n4) {
    int i = blockIdx.x * 256 + threadIdx.x;
    if (i >= n4) return;
    float4 f = ((const float4*)src)[i];
    ushort4 o; o.x = f2bf(f.x); o.y = f2bf(f.y); o.z = f2bf(f.z); o.w = f2bf(f.w);
    ((ushort4*)dst)[i] = o;
}

__global__ __launch_bounds__(256) void transpose_cvt_kernel(const float* __restrict__ W,
                                                            u16* __restrict__ WT) {
    __shared__ float t[32][33];
    int bx = blockIdx.x * 32, by = blockIdx.y * 32;
    int tx = threadIdx.x, ty = threadIdx.y;          // block 32x8
    for (int i = 0; i < 32; i += 8)
        t[ty + i][tx] = W[(size_t)(by + ty + i) * BDIM + bx + tx];
    __syncthreads();
    for (int i = 0; i < 32; i += 8)
        WT[(size_t)(bx + ty + i) * BDIM + by + tx] = f2bf(t[tx][ty + i]);
}

__global__ __launch_bounds__(256) void pack_bias_kernel(const float* bq, const float* bk,
                                                        const float* bv, float* out) {
    int i = blockIdx.x * 256 + threadIdx.x;
    if (i >= NQKV) return;
    out[i] = (i < 1024) ? bq[i] : (i < 2048) ? bk[i - 1024] : bv[i - 2048];
}

// ---------------------------------------------------------------- bf16 GEMM
// C[M][N] = A[M][K](bf16) * BT[N][K](bf16)^T + bias + residual
#define BM 128
#define BN 128
#define BK 32
#define LDT 40   // padded LDS row stride (halves): 80B, 16B aligned, conflict-free

__global__ __launch_bounds__(256) void gemm_bf16_kernel(
    const u16* __restrict__ A, const u16* __restrict__ BT,
    const float* __restrict__ bias, const float* __restrict__ residual,
    float* __restrict__ C, int M, int N, int K) {

    __shared__ u16 sA[2][BM * LDT];
    __shared__ u16 sB[2][BN * LDT];

    int bm = blockIdx.y * BM, bn = blockIdx.x * BN;
    int tid = threadIdx.x;
    int wave = tid >> 5, lane = tid & 31;
    int wm = wave & 1, wn = wave >> 1;     // 2x4 wave grid -> wave tile 64x32
    int laneRow = lane & 15, g = lane >> 4;

    // this thread's two staging chunks (row, col) within the 128x32 tile
    int c0 = tid * 2, c1 = tid * 2 + 1;
    int r0 = c0 >> 2, col0 = (c0 & 3) * 8;
    int r1 = c1 >> 2, col1 = (c1 & 3) * 8;

    v8f acc[4][2];
    for (int i = 0; i < 4; ++i)
        for (int j = 0; j < 2; ++j)
            acc[i][j] = v8f{0.f, 0.f, 0.f, 0.f, 0.f, 0.f, 0.f, 0.f};

    auto stage = [&](int buf, int k0) {
        async_copy16((u32)(uintptr_t)&sA[buf][r0 * LDT + col0],
                     A + (size_t)(bm + r0) * K + k0 + col0);
        async_copy16((u32)(uintptr_t)&sA[buf][r1 * LDT + col1],
                     A + (size_t)(bm + r1) * K + k0 + col1);
        async_copy16((u32)(uintptr_t)&sB[buf][r0 * LDT + col0],
                     BT + (size_t)(bn + r0) * K + k0 + col0);
        async_copy16((u32)(uintptr_t)&sB[buf][r1 * LDT + col1],
                     BT + (size_t)(bn + r1) * K + k0 + col1);
    };

    // prologue: fill buffer 0
    stage(0, 0);
    wait_async0();
    __syncthreads();

    int cur = 0;
    for (int k0 = 0; k0 < K; k0 += BK) {
        bool more = (k0 + BK) < K;
        if (more) stage(cur ^ 1, k0 + BK);   // overlap next-tile DMA with WMMA

        Frag bf[2];
        for (int nt = 0; nt < 2; ++nt)
            load_frag_u16(bf[nt], &sB[cur][(wn * 32 + nt * 16 + laneRow) * LDT], g);
        for (int mt = 0; mt < 4; ++mt) {
            Frag af;
            load_frag_u16(af, &sA[cur][(wm * 64 + mt * 16 + laneRow) * LDT], g);
            for (int nt = 0; nt < 2; ++nt)
                acc[mt][nt] = __builtin_amdgcn_wmma_f32_16x16x32_bf16(
                    false, af.v, false, bf[nt].v, (short)0, acc[mt][nt], false, false);
        }

        if (more) wait_async0();             // own async copies into cur^1 done
        __syncthreads();                     // everyone done reading cur / writing cur^1
        cur ^= 1;
    }

    for (int mt = 0; mt < 4; ++mt)
        for (int nt = 0; nt < 2; ++nt) {
            int col = bn + wn * 32 + nt * 16 + laneRow;
            float bv = bias ? bias[col] : 0.f;
            for (int r = 0; r < 8; ++r) {
                int row = bm + wm * 64 + mt * 16 + g * 8 + r;
                float v = acc[mt][nt][r] + bv;
                if (residual) v += residual[(size_t)row * N + col];
                C[(size_t)row * N + col] = v;
            }
        }
}

// ---------------------------------------------------------------- scale + rope
// qkv f32 [ROWS][3072] -> q_rot/k_rot bf16 [b,h,l,64], vT bf16 [b,h,64,l]
__global__ __launch_bounds__(128) void rope_kernel(
    const float* __restrict__ qkv, const float* __restrict__ phi,
    const float* __restrict__ mag, const float* __restrict__ gamma,
    u16* __restrict__ qrot, u16* __restrict__ krot, u16* __restrict__ vt) {

    int idx = blockIdx.x * 4 + (threadIdx.x >> 5);   // over B*H*L
    int lane = threadIdx.x & 31;
    int l = idx & (LDIM - 1);
    int h = (idx >> 10) & (HH - 1);
    int b = idx >> 14;

    float ph = phi[idx];
    float sc = 1.0f + gamma[h] * tanhf(mag[idx]);
    float cs = cosf(ph), sn = sinf(ph);

    const float* row = qkv + (size_t)(b * LDIM + l) * NQKV + h * HD;
    float q0 = row[lane] * sc,        q1 = row[lane + 32] * sc;
    float k0 = row[1024 + lane] * sc, k1 = row[1024 + lane + 32] * sc;
    float v0 = row[2048 + lane],      v1 = row[2048 + lane + 32];

    float qr0 = q0 * cs - q1 * sn, qr1 = q1 * cs + q0 * sn;
    float kr0 = k0 * cs - k1 * sn, kr1 = k1 * cs + k0 * sn;

    size_t ro = (size_t)idx * HD;
    qrot[ro + lane] = f2bf(qr0); qrot[ro + lane + 32] = f2bf(qr1);
    krot[ro + lane] = f2bf(kr0); krot[ro + lane + 32] = f2bf(kr1);

    size_t vb = (size_t)(b * HH + h) * HD;
    vt[(vb + lane) * LDIM + l]      = f2bf(v0);
    vt[(vb + lane + 32) * LDIM + l] = f2bf(v1);
}

// ---------------------------------------------------------------- attention
// one block per (b, h, 64-row query tile). Dynamic LDS: 64*1024 f32 scores.
__global__ __launch_bounds__(256) void attn_kernel(
    const u16* __restrict__ qrot, const u16* __restrict__ krot,
    const u16* __restrict__ vt, const float* __restrict__ mask,
    u16* __restrict__ ctx) {

    extern __shared__ float sc[];                    // [64][1024]
    int qt = blockIdx.x, h = blockIdx.y, b = blockIdx.z;
    int qbase = qt * 64;
    int tid = threadIdx.x, wave = tid >> 5, lane = tid & 31;
    int laneRow = lane & 15, g = lane >> 4;
    const float rscale = 0.125f;                     // 1/sqrt(64)

    const u16* qp = qrot + ((size_t)(b * HH + h) * LDIM + qbase) * HD;
    const u16* kp = krot + (size_t)(b * HH + h) * LDIM * HD;
    const u16* vp = vt + (size_t)(b * HH + h) * HD * LDIM;

    // ---- phase 1: scores = QK^T * rscale + mask ----
    Frag qf[4][2];
    for (int mt = 0; mt < 4; ++mt)
        for (int ks = 0; ks < 2; ++ks)
            load_frag_u16(qf[mt][ks], qp + (mt * 16 + laneRow) * HD + ks * 32, g);

    for (int nt = 0; nt < 8; ++nt) {
        int n = wave * 128 + nt * 16;                // key block
        Frag kf[2];
        for (int ks = 0; ks < 2; ++ks)
            load_frag_u16(kf[ks], kp + (n + laneRow) * HD + ks * 32, g);
        v8f c[4];
        for (int mt = 0; mt < 4; ++mt) {
            c[mt] = v8f{0.f, 0.f, 0.f, 0.f, 0.f, 0.f, 0.f, 0.f};
            for (int ks = 0; ks < 2; ++ks)
                c[mt] = __builtin_amdgcn_wmma_f32_16x16x32_bf16(
                    false, qf[mt][ks].v, false, kf[ks].v, (short)0, c[mt], false, false);
        }
        int col = n + laneRow;
        float mk = mask[b * LDIM + col];
        for (int mt = 0; mt < 4; ++mt)
            for (int r = 0; r < 8; ++r)
                sc[(mt * 16 + g * 8 + r) * LDIM + col] = c[mt][r] * rscale + mk;
    }
    __syncthreads();

    // ---- phase 2: softmax (+ diagonal fixup for fully-masked rows) ----
    for (int i = 0; i < 8; ++i) {
        int r = wave * 8 + i;
        float* row = sc + (size_t)r * LDIM;
        float mx = -3.4e38f;
        for (int j = lane; j < LDIM; j += 32) mx = fmaxf(mx, row[j]);
        for (int off = 16; off; off >>= 1) mx = fmaxf(mx, __shfl_xor(mx, off, 32));
        if (mx <= -1e8f) {
            if (lane == 0) row[qbase + r] = 0.f;
            mx = fmaxf(mx, 0.f);
        }
        float sum = 0.f;
        for (int j = lane; j < LDIM; j += 32) {
            float e = __expf(row[j] - mx);
            sum += e; row[j] = e;
        }
        for (int off = 16; off; off >>= 1) sum += __shfl_xor(sum, off, 32);
        float inv = 1.0f / sum;
        for (int j = lane; j < LDIM; j += 32) row[j] *= inv;
    }
    __syncthreads();

    // ---- phase 3: ctx = P @ V ----
    int mt = wave & 3;
    int nt0 = (wave >> 2) * 2;                       // waves 0-3 -> cols 0..31, 4-7 -> 32..63
    v8f c0 = v8f{0.f, 0.f, 0.f, 0.f, 0.f, 0.f, 0.f, 0.f};
    v8f c1 = c0;
    for (int k0 = 0; k0 < LDIM; k0 += 32) {
        // A fragment: probs f32 -> bf16 from LDS
        const float* prow = sc + (size_t)(mt * 16 + laneRow) * LDIM + k0;
        Frag af;
        {
            float4 f0 = *(const float4*)(prow + g * 8);
            float4 f1 = *(const float4*)(prow + g * 8 + 4);
            float4 f2 = *(const float4*)(prow + 16 + g * 8);
            float4 f3 = *(const float4*)(prow + 16 + g * 8 + 4);
            af.h[0] = f2bf(f0.x);  af.h[1] = f2bf(f0.y);
            af.h[2] = f2bf(f0.z);  af.h[3] = f2bf(f0.w);
            af.h[4] = f2bf(f1.x);  af.h[5] = f2bf(f1.y);
            af.h[6] = f2bf(f1.z);  af.h[7] = f2bf(f1.w);
            af.h[8] = f2bf(f2.x);  af.h[9] = f2bf(f2.y);
            af.h[10] = f2bf(f2.z); af.h[11] = f2bf(f2.w);
            af.h[12] = f2bf(f3.x); af.h[13] = f2bf(f3.y);
            af.h[14] = f2bf(f3.z); af.h[15] = f2bf(f3.w);
        }
        Frag b0, b1;
        load_frag_u16(b0, vp + (size_t)(nt0 * 16 + laneRow) * LDIM + k0, g);
        load_frag_u16(b1, vp + (size_t)((nt0 + 1) * 16 + laneRow) * LDIM + k0, g);
        c0 = __builtin_amdgcn_wmma_f32_16x16x32_bf16(false, af.v, false, b0.v, (short)0, c0, false, false);
        c1 = __builtin_amdgcn_wmma_f32_16x16x32_bf16(false, af.v, false, b1.v, (short)0, c1, false, false);
    }
    for (int p = 0; p < 2; ++p) {
        v8f cc = p ? c1 : c0;
        int col = h * HD + (nt0 + p) * 16 + laneRow;
        for (int r = 0; r < 8; ++r) {
            int l = qbase + mt * 16 + g * 8 + r;
            ctx[((size_t)b * LDIM + l) * BDIM + col] = f2bf(cc[r]);
        }
    }
}

// ---------------------------------------------------------------- LayerNorm
__global__ __launch_bounds__(256) void ln_kernel(const float* __restrict__ y,
                                                 const float* __restrict__ lnw,
                                                 const float* __restrict__ lnb,
                                                 float* __restrict__ out) {
    __shared__ float rbuf[16];
    int row = blockIdx.x, tid = threadIdx.x;
    int wave = tid >> 5, lane = tid & 31;
    const float* x = y + (size_t)row * BDIM;
    float s = 0.f, s2 = 0.f;
    for (int j = tid; j < BDIM; j += 256) { float v = x[j]; s += v; s2 += v * v; }
    for (int off = 16; off; off >>= 1) {
        s += __shfl_xor(s, off, 32);
        s2 += __shfl_xor(s2, off, 32);
    }
    if (lane == 0) { rbuf[wave] = s; rbuf[8 + wave] = s2; }
    __syncthreads();
    float S = 0.f, S2 = 0.f;
    for (int i = 0; i < 8; ++i) { S += rbuf[i]; S2 += rbuf[8 + i]; }
    float mu = S * (1.0f / BDIM);
    float var = S2 * (1.0f / BDIM) - mu * mu;
    float inv = rsqrtf(var + 1e-12f);
    for (int j = tid; j < BDIM; j += 256)
        out[(size_t)row * BDIM + j] = (x[j] - mu) * inv * lnw[j] + lnb[j];
}

// ---------------------------------------------------------------- launch
extern "C" void kernel_launch(void* const* d_in, const int* in_sizes, int n_in,
                              void* d_out, int out_size, void* d_ws, size_t ws_size,
                              hipStream_t stream) {
    const float* hidden = (const float*)d_in[0];
    const float* mask   = (const float*)d_in[1];
    const float* phi    = (const float*)d_in[2];
    const float* mag    = (const float*)d_in[3];
    const float* Wq     = (const float*)d_in[4];
    const float* bq     = (const float*)d_in[5];
    const float* Wk     = (const float*)d_in[6];
    const float* bk     = (const float*)d_in[7];
    const float* Wv     = (const float*)d_in[8];
    const float* bv     = (const float*)d_in[9];
    const float* gamma  = (const float*)d_in[10];
    const float* Wo     = (const float*)d_in[11];
    const float* bo     = (const float*)d_in[12];
    const float* lnw    = (const float*)d_in[13];
    const float* lnb    = (const float*)d_in[14];
    float* out = (float*)d_out;

    char* ws = (char*)d_ws;
    const size_t o_xb   = 0;                           // 8 MB  bf16 X
    const size_t o_wt   = o_xb + 8388608ull;           // 6 MB  bf16 WT (q,k,v packed)
    const size_t o_wot  = o_wt + 6291456ull;           // 2 MB  bf16 WoT
    const size_t o_bias = o_wot + 2097152ull;          // 12 KB packed bias
    const size_t o_qkv  = o_bias + 12288ull;           // 48 MB f32 qkv (reused for y)
    const size_t o_qr   = o_qkv + 50331648ull;         // 8 MB
    const size_t o_kr   = o_qr + 8388608ull;           // 8 MB
    const size_t o_vt   = o_kr + 8388608ull;           // 8 MB
    const size_t o_ctx  = o_vt + 8388608ull;           // 8 MB
    const size_t o_y    = o_qkv;                       // reuse dead region

    u16*   Xb    = (u16*)(ws + o_xb);
    u16*   WT    = (u16*)(ws + o_wt);
    u16*   WoT   = (u16*)(ws + o_wot);
    float* biasA = (float*)(ws + o_bias);
    float* qkv   = (float*)(ws + o_qkv);
    u16*   qrot  = (u16*)(ws + o_qr);
    u16*   krot  = (u16*)(ws + o_kr);
    u16*   vtb   = (u16*)(ws + o_vt);
    u16*   ctxb  = (u16*)(ws + o_ctx);
    float* yb    = (float*)(ws + o_y);

    // 1) convert X to bf16
    cvt_x_kernel<<<4096, 256, 0, stream>>>(hidden, Xb, ROWS * BDIM / 4);

    // 2) transpose-convert weights
    dim3 tb(32, 8), tg(32, 32);
    transpose_cvt_kernel<<<tg, tb, 0, stream>>>(Wq, WT);
    transpose_cvt_kernel<<<tg, tb, 0, stream>>>(Wk, WT + (size_t)BDIM * BDIM);
    transpose_cvt_kernel<<<tg, tb, 0, stream>>>(Wv, WT + 2ull * BDIM * BDIM);
    transpose_cvt_kernel<<<tg, tb, 0, stream>>>(Wo, WoT);
    pack_bias_kernel<<<12, 256, 0, stream>>>(bq, bk, bv, biasA);

    // 3) fused QKV GEMM: [4096 x 3072]
    gemm_bf16_kernel<<<dim3(NQKV / BN, ROWS / BM), 256, 0, stream>>>(
        Xb, WT, biasA, nullptr, qkv, ROWS, NQKV, BDIM);

    // 4) scale + rotation + V transpose
    rope_kernel<<<BB * HH * LDIM / 4, 128, 0, stream>>>(
        qkv, phi, mag, gamma, qrot, krot, vtb);

    // 5) fused attention (scores strip in 256KB LDS)
    static_assert(64 * LDIM * sizeof(float) == 262144, "lds");
    hipFuncSetAttribute((const void*)attn_kernel,
                        hipFuncAttributeMaxDynamicSharedMemorySize, 262144);
    attn_kernel<<<dim3(LDIM / 64, HH, BB), 256, 262144, stream>>>(
        qrot, krot, vtb, mask, ctxb);

    // 6) output projection + residual
    gemm_bf16_kernel<<<dim3(BDIM / BN, ROWS / BM), 256, 0, stream>>>(
        ctxb, WoT, bo, hidden, yb, ROWS, BDIM, BDIM);

    // 7) LayerNorm
    ln_kernel<<<ROWS, 256, 0, stream>>>(yb, lnw, lnb, out);
}
